// MultilingualContrastiveLoss_26929444946428
// MI455X (gfx1250) — matrix operand order repaired and checked by reading, not compiled, not run on hardware
//
#include <hip/hip_runtime.h>

typedef __attribute__((ext_vector_type(16))) _Float16 v16h;
typedef __attribute__((ext_vector_type(8)))  _Float16 v8h;
typedef __attribute__((ext_vector_type(8)))  float    v8f;

#define BB 4096
#define DD 1024
#define NEG_INF (-__builtin_huge_valf())

// ---------------------------------------------------------------------------
// Zero the 9 per-matrix loss accumulators.
// ---------------------------------------------------------------------------
__global__ void zero_accum_kernel(float* p) {
    if (threadIdx.x < 9) p[threadIdx.x] = 0.0f;
}

// ---------------------------------------------------------------------------
// Row-wise L2 normalize (eps = 1e-12 clamp, matching F.normalize) and convert
// to f16 for the WMMA operands. One block per row; 4 matrices stacked in grid.
// ---------------------------------------------------------------------------
__global__ void normalize_f16_kernel(const float* __restrict__ v,
                                     const float* __restrict__ e,
                                     const float* __restrict__ f,
                                     const float* __restrict__ d,
                                     _Float16* __restrict__ nv,
                                     _Float16* __restrict__ ne,
                                     _Float16* __restrict__ nf,
                                     _Float16* __restrict__ nd) {
    const int gid = blockIdx.x;
    const int mat = gid >> 12;          // / 4096
    const int row = gid & (BB - 1);
    const int tid = threadIdx.x;        // 256 threads

    const float* src;
    _Float16* dst;
    if      (mat == 0) { src = v; dst = nv; }
    else if (mat == 1) { src = e; dst = ne; }
    else if (mat == 2) { src = f; dst = nf; }
    else               { src = d; dst = nd; }

    const float* r = src + (size_t)row * DD;
    float ss = 0.0f;
    #pragma unroll
    for (int c = tid; c < DD; c += 256) { float x = r[c]; ss += x * x; }

    __shared__ float red[256];
    red[tid] = ss;
    __syncthreads();
    for (int s = 128; s > 0; s >>= 1) {
        if (tid < s) red[tid] += red[tid + s];
        __syncthreads();
    }
    const float scale = 1.0f / fmaxf(sqrtf(red[0]), 1e-12f);

    _Float16* o = dst + (size_t)row * DD;
    #pragma unroll
    for (int c = tid; c < DD; c += 256) o[c] = (_Float16)(r[c] * scale);
}

// ---------------------------------------------------------------------------
// S[4096x4096] f32 = Xn[4096x1024] f16 @ Yn[4096x1024]^T f16
// Block = 256 threads (8 waves), 128(M) x 256(N) tile; wave = 64x64
// (4x4 wmma tiles) -> 16 WMMAs per 16 b128 loads per K-step of 32.
// A-fragment lane layout (16x32 f16): lane m (0-15) / m+16 hold
//   K {hi*8..+7, 16+hi*8..+7} of row m -> two contiguous b128 loads.
// B-fragment lane layout (32x16 f16): lane n / n+16 hold K {hi*8..+7,
//   16+hi*8..+7} of row n of Y -> two contiguous b128 loads.
// ---------------------------------------------------------------------------
__device__ __forceinline__ v16h pack16(const _Float16* p0, const _Float16* p1) {
    v8h a = *(const v8h*)p0;
    v8h b = *(const v8h*)p1;
    return __builtin_shufflevector(a, b, 0,1,2,3,4,5,6,7,8,9,10,11,12,13,14,15);
}

__global__ void __launch_bounds__(256)
gemm_xyT_wmma_kernel(const _Float16* __restrict__ A,
                     const _Float16* __restrict__ B,
                     float* __restrict__ S) {
    const int tid    = threadIdx.x & 31;
    const int wave   = threadIdx.x >> 5;   // 0..7
    const int waveM  = wave >> 2;          // 0..1
    const int waveN  = wave & 3;           // 0..3
    const int m_base = blockIdx.y * 128 + waveM * 64;
    const int n_base = blockIdx.x * 256 + waveN * 64;
    const int lane16 = tid & 15;
    const int hi     = tid >> 4;

    const _Float16* arow[4];
    const _Float16* brow[4];
    #pragma unroll
    for (int i = 0; i < 4; ++i) arow[i] = A + (size_t)(m_base + i * 16 + lane16) * DD;
    #pragma unroll
    for (int j = 0; j < 4; ++j) brow[j] = B + (size_t)(n_base + j * 16 + lane16) * DD;

    v8f acc[4][4];
    #pragma unroll
    for (int i = 0; i < 4; ++i)
        #pragma unroll
        for (int j = 0; j < 4; ++j) acc[i][j] = (v8f){0,0,0,0,0,0,0,0};

    for (int k0 = 0; k0 < DD; k0 += 32) {
        // Speculative prefetch of the next K-slab (dropped harmlessly at end).
        __builtin_prefetch(arow[0] + k0 + 32, 0, 1);
        __builtin_prefetch(brow[0] + k0 + 32, 0, 1);

        v16h af[4], bf[4];
        #pragma unroll
        for (int i = 0; i < 4; ++i)
            af[i] = pack16(arow[i] + k0 + hi * 8, arow[i] + k0 + 16 + hi * 8);
        #pragma unroll
        for (int j = 0; j < 4; ++j)
            bf[j] = pack16(brow[j] + k0 + hi * 8, brow[j] + k0 + 16 + hi * 8);

        #pragma unroll
        for (int i = 0; i < 4; ++i)
            #pragma unroll
            for (int j = 0; j < 4; ++j)
                acc[i][j] = __builtin_amdgcn_wmma_f32_16x16x32_f16(
                    false, af[i], false, bf[j], (short)0, acc[i][j], false, false);
    }

    // D f32 16x16 layout: VGPR r -> row hi*8 + r, lane&15 -> column.
    #pragma unroll
    for (int j = 0; j < 4; ++j) {
        const int c = n_base + j * 16 + lane16;
        #pragma unroll
        for (int i = 0; i < 4; ++i) {
            const int mrow = m_base + i * 16 + hi * 8;
            #pragma unroll
            for (int r = 0; r < 8; ++r)
                S[(size_t)(mrow + r) * BB + c] = acc[i][j][r];
        }
    }
}

// ---------------------------------------------------------------------------
// Per-row pair loss: top-10 off-diagonal + streaming logsumexp([pos, hn]).
// One block (256 threads) per row; accumulates row loss into *accum.
// ---------------------------------------------------------------------------
__global__ void __launch_bounds__(256)
pair_loss_kernel(const float* __restrict__ S, float* __restrict__ accum) {
    const int row = blockIdx.x;
    const int tid = threadIdx.x;
    const float* r = S + (size_t)row * BB;

    // Private sorted (descending) top-10 — static indices keep it in VGPRs.
    float top[10];
    #pragma unroll
    for (int t = 0; t < 10; ++t) top[t] = NEG_INF;

    #pragma unroll
    for (int c = tid; c < BB; c += 256) {
        if (c == row) continue;
        float x = r[c];
        if (x > top[9]) {
            #pragma unroll
            for (int t = 0; t < 10; ++t) {
                if (x > top[t]) { float tmp = top[t]; top[t] = x; x = tmp; }
            }
        }
    }

    __shared__ float smax[256];
    __shared__ int   sidx[256];

    float m0 = 0.0f, sexp = 0.0f;   // running logsumexp of selected negatives
    #pragma unroll
    for (int t = 0; t < 10; ++t) {
        smax[tid] = top[0];
        sidx[tid] = tid;
        __syncthreads();
        for (int s = 128; s > 0; s >>= 1) {
            if (tid < s) {
                if (smax[tid + s] > smax[tid]) {
                    smax[tid] = smax[tid + s];
                    sidx[tid] = sidx[tid + s];
                }
            }
            __syncthreads();
        }
        const float w    = smax[0];
        const int winner = sidx[0];
        __syncthreads();
        if (tid == winner) {            // pop head of winner's list
            #pragma unroll
            for (int q = 0; q < 9; ++q) top[q] = top[q + 1];
            top[9] = NEG_INF;
        }
        if (t == 0) { m0 = w; sexp = 1.0f; }
        else        { sexp += __expf(w - m0); }
    }

    if (tid == 0) {
        const float pos = r[row];
        const float m   = fmaxf(pos, m0);
        const float lse = m + __logf(__expf(pos - m) + sexp * __expf(m0 - m));
        atomicAdd(accum, -pos + lse + 0.5f);   // MARGIN = 0.5
    }
}

// ---------------------------------------------------------------------------
// Final weighted combination of the 9 mean losses with temps, / 6.
// pl: [v.en, v.fr, v.de, en.fr, fr.en, en.de, de.en, fr.de, de.fr] (row sums)
// ---------------------------------------------------------------------------
__global__ void combine_kernel(const float* __restrict__ pl,
                               const float* __restrict__ temps,
                               float* __restrict__ out) {
    if (threadIdx.x == 0 && blockIdx.x == 0) {
        const float invB = 1.0f / (float)BB;
        const float a    = pl[0] * invB;   // pl(sim(video,en))
        const float b    = pl[1] * invB;   // pl(sim(video,fr))
        const float c    = pl[2] * invB;   // pl(sim(video,de))
        const float p01  = pl[3] * invB;   // pl(sim(en,fr))
        const float p01T = pl[4] * invB;   // pl(sim(fr,en))
        const float p02  = pl[5] * invB;   // pl(sim(en,de))
        const float p02T = pl[6] * invB;   // pl(sim(de,en))
        const float p12  = pl[7] * invB;   // pl(sim(fr,de))
        const float p12T = pl[8] * invB;   // pl(sim(de,fr))
        const float total =
              (a + p01 ) * temps[0] + (a + p02 ) * temps[1]
            + (b + p01T) * temps[2] + (b + p12 ) * temps[3]
            + (c + p02T) * temps[4] + (c + p12T) * temps[5];
        out[0] = total / 6.0f;
    }
}

// ---------------------------------------------------------------------------
extern "C" void kernel_launch(void* const* d_in, const int* in_sizes, int n_in,
                              void* d_out, int out_size, void* d_ws, size_t ws_size,
                              hipStream_t stream) {
    const float* video = (const float*)d_in[0];
    const float* t_en  = (const float*)d_in[1];
    const float* t_fr  = (const float*)d_in[2];
    const float* t_de  = (const float*)d_in[3];
    const float* temps = (const float*)d_in[4];

    // Workspace layout:
    //   4 x (4096*1024 f16) normalized operands  = 32 MB
    //   1 x (4096*4096 f32) similarity buffer    = 64 MB (reused 9x)
    //   9 x f32 loss accumulators
    char* ws = (char*)d_ws;
    const size_t NRM_BYTES = (size_t)BB * DD * sizeof(_Float16);   // 8 MB
    const size_t S_BYTES   = (size_t)BB * BB * sizeof(float);      // 64 MB
    _Float16* nrm[4];
    for (int i = 0; i < 4; ++i) nrm[i] = (_Float16*)(ws + i * NRM_BYTES);
    float* S  = (float*)(ws + 4 * NRM_BYTES);
    float* pl = (float*)(ws + 4 * NRM_BYTES + S_BYTES);

    zero_accum_kernel<<<1, 32, 0, stream>>>(pl);

    normalize_f16_kernel<<<4 * BB, 256, 0, stream>>>(
        video, t_en, t_fr, t_de, nrm[0], nrm[1], nrm[2], nrm[3]);

    // 0=video 1=en 2=fr 3=de; ordered (X,Y) for S = sim(X, Y)
    const int pa[9] = {0, 0, 0, 1, 2, 1, 3, 2, 3};
    const int pb[9] = {1, 2, 3, 2, 1, 3, 1, 3, 2};
    const dim3 ggrid(BB / 256, BB / 128);   // 16 x 32
    for (int i = 0; i < 9; ++i) {
        gemm_xyT_wmma_kernel<<<ggrid, 256, 0, stream>>>(nrm[pa[i]], nrm[pb[i]], S);
        pair_loss_kernel<<<BB, 256, 0, stream>>>(S, pl + i);
    }

    combine_kernel<<<1, 1, 0, stream>>>(pl, temps, (float*)d_out);
}
